// BitNetAttention_44427141710297
// MI455X (gfx1250) — matrix operand order repaired and checked by reading, not compiled, not run on hardware
//
#include <hip/hip_runtime.h>
#include <hip/hip_bf16.h>
#include <math.h>

typedef _Float16 v16h __attribute__((ext_vector_type(16)));
typedef _Float16 v8h  __attribute__((ext_vector_type(8)));
typedef float    v8f  __attribute__((ext_vector_type(8)));

#define HIDDEN 2048
#define NH     16
#define HD     128
#define SEQ    2048
#define NOUT   (3*HIDDEN)
#define SM_SCALE 0.08838834764831845f  /* 1/sqrt(128) */

// ---------------------------------------------------------------------------
// CDNA5 async global->LDS copy (ASYNCcnt-tracked, no VGPR round trip).
// LDS operand: low 32 bits of the flat shared address == wave LDS offset
// (ISA 10.2 aperture mapping).
// ---------------------------------------------------------------------------
__device__ inline void async_ld_b128(void* lds, const void* gsrc) {
  asm volatile("global_load_async_to_lds_b128 %0, %1, off"
               :: "v"((unsigned)(unsigned long long)lds), "v"(gsrc)
               : "memory");
}
__device__ inline void wait_async0() {
  asm volatile("s_wait_asynccnt 0" ::: "memory");
}

// ---------------------------------------------------------------------------
// WMMA helper: D = A(16x32 f16) * B(32x16 f16) + C(16x16 f32)
// ---------------------------------------------------------------------------
__device__ inline v8f wmma_f16(v16h a, v16h b, v8f c) {
  return __builtin_amdgcn_wmma_f32_16x16x32_f16(
      /*neg_a=*/false, a, /*neg_b=*/false, b,
      /*c_mod=*/(short)0, c, /*reuse_a=*/false, /*reuse_b=*/false);
}

// A-fragment (16x32, 16-bit): lane m=lane%16 holds row m; halves cover
// K = [8g,8g+8) and [16+8g,24+8g) where g = lane/16  (ISA 7.12.2)
__device__ inline v16h load_frag_a(const _Float16* tile, int row, int stride, int lane) {
  const int m = lane & 15, g = lane >> 4;
  const _Float16* p = tile + (size_t)(row + m) * stride;
  v8h x0 = *reinterpret_cast<const v8h*>(p + 8 * g);
  v8h x1 = *reinterpret_cast<const v8h*>(p + 16 + 8 * g);
  v16h r;
#pragma unroll
  for (int i = 0; i < 8; ++i) { r[i] = x0[i]; r[i + 8] = x1[i]; }
  return r;
}

// B-fragment (32x16, 16-bit): lane n=lane%16 holds col n; halves cover
// K = [8g,8g+8) and [16+8g,24+8g) where g = lane/16. Source tile stored [col][K] row-major.
__device__ inline v16h load_frag_b(const _Float16* tile, int col, int kof, int stride, int lane) {
  const int n = lane & 15, g = lane >> 4;
  const _Float16* p = tile + (size_t)(col + n) * stride + kof;
  v8h x0 = *reinterpret_cast<const v8h*>(p + 8 * g);
  v8h x1 = *reinterpret_cast<const v8h*>(p + 16 + 8 * g);
  v16h r;
#pragma unroll
  for (int i = 0; i < 8; ++i) { r[i] = x0[i]; r[i + 8] = x1[i]; }
  return r;
}

// ---------------------------------------------------------------------------
// Scale computation (BitNet absmean)
// ---------------------------------------------------------------------------
__global__ void init_red(float* red) {
  if (threadIdx.x < 8) red[threadIdx.x] = 0.0f;
}

__global__ __launch_bounds__(256) void absmean_reduce(const float* __restrict__ x,
                                                      int n, float* accum) {
  __shared__ float sm[256];
  float s = 0.0f;
  for (int i = blockIdx.x * blockDim.x + threadIdx.x; i < n; i += gridDim.x * blockDim.x)
    s += fabsf(x[i]);
  sm[threadIdx.x] = s;
  __syncthreads();
  for (int o = 128; o > 0; o >>= 1) {
    if ((int)threadIdx.x < o) sm[threadIdx.x] += sm[threadIdx.x + o];
    __syncthreads();
  }
  if (threadIdx.x == 0) atomicAdd(accum, sm[0]);
}

__global__ void finalize_scales(float* red) {
  if (threadIdx.x == 0) {
    float sp = red[0] / (float)((size_t)NOUT * HIDDEN) + 1e-5f;
    float so = red[1] / (float)((size_t)HIDDEN * HIDDEN) + 1e-5f;
    red[2] = sp; red[3] = 1.0f / sp;
    red[4] = so; red[5] = 1.0f / so;
  }
}

// ---------------------------------------------------------------------------
// Tiled GEMM: out[s][o] = sum_h A[s][h] * ternary(W[o][h]) * scale
// KIND 0: A = hidden_states (f32), epilogue scatters f16 into q/k (head-major
//         [h][s][d]) and v TRANSPOSED ([h][d][s]) for async attention staging.
// KIND 1: A = ctx (f16, async-staged), epilogue writes f32 to d_out.
// Block 256 thr = 8 waves; tile 128x128; K-tile 32; wave = 2x4 WMMA frags.
// ---------------------------------------------------------------------------
template <int KIND>
__global__ __launch_bounds__(256) void gemm_ws(const float* __restrict__ Af32,
                                               const _Float16* __restrict__ Af16,
                                               const float* __restrict__ W,
                                               const float* __restrict__ sc,
                                               _Float16* qg, _Float16* kg, _Float16* vgT,
                                               float* out) {
  __shared__ __align__(16) _Float16 At[128 * 32];
  __shared__ __align__(16) _Float16 Bt[128 * 32];
  const int tid = threadIdx.x;
  const int lane = tid & 31, wid = tid >> 5;
  const int wm = wid & 3, wn = wid >> 2;      // 4x2 wave grid
  const int bm = blockIdx.x * 128;            // S rows
  const int bn = blockIdx.y * 128;            // O cols
  const float s_w = sc[0];
  const float inv_s = sc[1];

  v8f acc[2][4];
#pragma unroll
  for (int i = 0; i < 2; ++i)
#pragma unroll
    for (int j = 0; j < 4; ++j) acc[i][j] = {};

  const int ar = tid >> 1;            // staging row 0..127
  const int ac = (tid & 1) * 16;      // staging col block

  for (int kk = 0; kk < HIDDEN / 32; ++kk) {
    const int k0 = kk * 32;
    // stage A tile [128 x 32]
    if constexpr (KIND == 0) {
      // convert f32 -> f16 through VGPRs
      const float* src = Af32 + (size_t)(bm + ar) * HIDDEN + k0 + ac;
#pragma unroll
      for (int i = 0; i < 16; ++i) At[ar * 32 + ac + i] = (_Float16)src[i];
      if (kk + 1 < HIDDEN / 32) __builtin_prefetch(src + 32, 0, 1);
    } else {
      // straight f16 copy: async global->LDS (2 x b128 per thread)
#pragma unroll
      for (int c = 0; c < 2; ++c) {
        const int idx = tid * 2 + c;            // 0..511
        const int row = idx >> 2, part = idx & 3;
        async_ld_b128(At + row * 32 + part * 8,
                      Af16 + (size_t)(bm + row) * HIDDEN + k0 + part * 8);
      }
    }
    // stage B tile [128 cols x 32 K]: quantize to ternary f16 on the fly
    {
      const float* src = W + (size_t)(bn + ar) * HIDDEN + k0 + ac;
#pragma unroll
      for (int i = 0; i < 16; ++i) {
        float q = rintf(src[i] * inv_s);            // round-half-even
        q = fminf(1.0f, fmaxf(-1.0f, q));
        Bt[ar * 32 + ac + i] = (_Float16)q;
      }
      if (kk + 1 < HIDDEN / 32) __builtin_prefetch(src + 32, 0, 1);
    }
    if constexpr (KIND == 1) wait_async0();
    __syncthreads();

    v16h a0 = load_frag_a(At, wm * 32, 32, lane);
    v16h a1 = load_frag_a(At, wm * 32 + 16, 32, lane);
#pragma unroll
    for (int nf = 0; nf < 4; ++nf) {
      v16h b = load_frag_b(Bt, wn * 64 + nf * 16, 0, 32, lane);
      acc[0][nf] = wmma_f16(a0, b, acc[0][nf]);
      acc[1][nf] = wmma_f16(a1, b, acc[1][nf]);
    }
    __syncthreads();
  }

  // Epilogue (D layout: lane holds col n=lane%16; VGPR r -> row r + 8*(lane/16))
  const int gq = lane >> 4, nn = lane & 15;
#pragma unroll
  for (int mf = 0; mf < 2; ++mf)
#pragma unroll
    for (int nf = 0; nf < 4; ++nf)
#pragma unroll
      for (int r = 0; r < 8; ++r) {
        const float v = acc[mf][nf][r] * s_w;
        const int row = bm + wm * 32 + mf * 16 + r + 8 * gq;
        const int col = bn + wn * 64 + nf * 16 + nn;
        if constexpr (KIND == 0) {
          const int which = col >> 11;      // 0:q 1:k 2:v
          const int w2 = col & 2047;
          const int head = w2 >> 7, d = w2 & 127;
          if (which == 0) {
            qg[((size_t)head * SEQ + row) * HD + d] = (_Float16)v;
          } else if (which == 1) {
            kg[((size_t)head * SEQ + row) * HD + d] = (_Float16)v;
          } else {
            // V stored transposed: [h][d][s]
            vgT[((size_t)head * HD + d) * SEQ + row] = (_Float16)v;
          }
        } else {
          out[(size_t)row * HIDDEN + col] = v;
        }
      }
}

// ---------------------------------------------------------------------------
// RoPE on q and k (head-major [nh][S][hd], f16). Thread i handles (d=i, d=i+64).
// ---------------------------------------------------------------------------
__global__ void rope_kernel(_Float16* qg, _Float16* kg) {
  const int h = blockIdx.x, s = blockIdx.y, i = threadIdx.x;  // i in [0,64)
  // inv_freq = 10000^(-i/64) ; ln(10000) = 9.210340371976184
  const float ang = (float)s * __expf(-(float)i * (9.210340371976184f / 64.0f));
  float sn, cs;
  __sincosf(ang, &sn, &cs);
  const size_t base = ((size_t)h * SEQ + s) * HD;

  float q1 = (float)qg[base + i], q2 = (float)qg[base + i + 64];
  qg[base + i]      = (_Float16)(q1 * cs - q2 * sn);
  qg[base + i + 64] = (_Float16)(q2 * cs + q1 * sn);

  float k1 = (float)kg[base + i], k2 = (float)kg[base + i + 64];
  kg[base + i]      = (_Float16)(k1 * cs - k2 * sn);
  kg[base + i + 64] = (_Float16)(k2 * cs + k1 * sn);
}

// ---------------------------------------------------------------------------
// Flash attention: block = (head, 128 q rows); 8 waves x 16 rows.
// KV chunks of 32 async-staged in LDS (V already transposed in global).
// Online softmax in WMMA fragment layout.
// ---------------------------------------------------------------------------
__global__ __launch_bounds__(256) void attn_kernel(const _Float16* __restrict__ qg,
                                                   const _Float16* __restrict__ kg,
                                                   const _Float16* __restrict__ vgT,
                                                   _Float16* __restrict__ ctxg) {
  __shared__ __align__(16) _Float16 kt[32 * 128];    // [kv][d]
  __shared__ __align__(16) _Float16 vtT[128 * 32];   // [d][kv]
  __shared__ __align__(16) _Float16 pt[8 * 16 * 32]; // per-wave probs [16][32]
  const int h = blockIdx.x, qt = blockIdx.y;
  const int tid = threadIdx.x, lane = tid & 31, wid = tid >> 5;
  const int qb = qt * 128 + wid * 16;                // this wave's q-row base
  const int gq = lane >> 4, nn = lane & 15;
  const _Float16* qhead  = qg  + (size_t)h * SEQ * HD;
  const _Float16* khead  = kg  + (size_t)h * SEQ * HD;
  const _Float16* vheadT = vgT + (size_t)h * HD * SEQ;   // [d][s]
  _Float16* myp = pt + wid * (16 * 32);

  // Load q A-fragments once: 4 frags cover hd=128
  v16h qf[4];
#pragma unroll
  for (int kf = 0; kf < 4; ++kf)
    qf[kf] = load_frag_a(qhead + (size_t)qb * HD + kf * 32, 0, HD, lane);

  v8f ctx[8];
#pragma unroll
  for (int df = 0; df < 8; ++df) ctx[df] = {};
  float mrow[8], lrow[8];
#pragma unroll
  for (int r = 0; r < 8; ++r) { mrow[r] = -1e30f; lrow[r] = 0.0f; }

  const int nchunks = qt * 4 + 4;  // causal: only kv <= block's max q row

  for (int kc = 0; kc < nchunks; ++kc) {
    const int kvb = kc * 32;
    // stage K chunk [32 kv][128 d]: straight async copy (2 x b128 per thread)
#pragma unroll
    for (int c = 0; c < 2; ++c) {
      const int idx = tid * 2 + c;              // 0..511
      const int r = idx >> 4, part = idx & 15;  // 16 b128 per kv row
      async_ld_b128(kt + r * 128 + part * 8,
                    khead + (size_t)(kvb + r) * HD + part * 8);
    }
    // stage V chunk [128 d][32 kv]: straight async copy from transposed global
#pragma unroll
    for (int c = 0; c < 2; ++c) {
      const int idx = tid * 2 + c;              // 0..511
      const int d = idx >> 2, part = idx & 3;   // 4 b128 per d row
      async_ld_b128(vtT + d * 32 + part * 8,
                    vheadT + (size_t)d * SEQ + kvb + part * 8);
    }
    wait_async0();
    __syncthreads();

    // scores [16 x 32] = q . k^T  (K dim = hd = 128 -> 4 WMMAs per 16-col frag)
    v8f sf[2];
    sf[0] = {}; sf[1] = {};
#pragma unroll
    for (int nf = 0; nf < 2; ++nf)
#pragma unroll
      for (int kf = 0; kf < 4; ++kf) {
        v16h b = load_frag_b(kt, nf * 16, kf * 32, 128, lane);
        sf[nf] = wmma_f16(qf[kf], b, sf[nf]);
      }

    // scale + causal mask + online softmax (row = r + 8*gq; col = kvb + frag*16 + nn)
    float rmax[8];
#pragma unroll
    for (int r = 0; r < 8; ++r) {
      const int qrow = qb + r + 8 * gq;
      float s0 = sf[0][r] * SM_SCALE;
      float s1 = sf[1][r] * SM_SCALE;
      if (kvb + nn > qrow) s0 = -1e30f;
      if (kvb + 16 + nn > qrow) s1 = -1e30f;
      sf[0][r] = s0; sf[1][r] = s1;
      float m = fmaxf(s0, s1);
#pragma unroll
      for (int x = 1; x < 16; x <<= 1) m = fmaxf(m, __shfl_xor(m, x, 32));
      rmax[r] = m;
    }
#pragma unroll
    for (int r = 0; r < 8; ++r) {
      const float mnew = fmaxf(mrow[r], rmax[r]);
      const float alpha = __expf(mrow[r] - mnew);
      mrow[r] = mnew;
      const float p0 = (sf[0][r] <= -1e29f) ? 0.0f : __expf(sf[0][r] - mnew);
      const float p1 = (sf[1][r] <= -1e29f) ? 0.0f : __expf(sf[1][r] - mnew);
      float rs = p0 + p1;
#pragma unroll
      for (int x = 1; x < 16; x <<= 1) rs += __shfl_xor(rs, x, 32);
      lrow[r] = lrow[r] * alpha + rs;
#pragma unroll
      for (int df = 0; df < 8; ++df) ctx[df][r] *= alpha;
      const int mloc = r + 8 * gq;
      myp[mloc * 32 + nn] = (_Float16)p0;
      myp[mloc * 32 + 16 + nn] = (_Float16)p1;
    }
    // wave-local LDS RAW: make sure probs stores land before A-frag reads
    asm volatile("s_wait_dscnt 0" ::: "memory");

    // ctx += P[16x32] . V[32x128]  (8 WMMAs)
    v16h pa = load_frag_a(myp, 0, 32, lane);
#pragma unroll
    for (int df = 0; df < 8; ++df) {
      v16h b = load_frag_b(vtT, df * 16, 0, 32, lane);
      ctx[df] = wmma_f16(pa, b, ctx[df]);
    }
    __syncthreads();  // protect kt/vtT before restage (uniform trip count)
  }

  // normalize and write ctx (f16) into [s][H] layout for the O GEMM
#pragma unroll
  for (int df = 0; df < 8; ++df)
#pragma unroll
    for (int r = 0; r < 8; ++r) {
      const int qrow = qb + r + 8 * gq;
      const float v = ctx[df][r] / lrow[r];
      ctxg[(size_t)qrow * HIDDEN + h * HD + df * 16 + nn] = (_Float16)v;
    }
}

// ---------------------------------------------------------------------------
// Launch
// ---------------------------------------------------------------------------
extern "C" void kernel_launch(void* const* d_in, const int* in_sizes, int n_in,
                              void* d_out, int out_size, void* d_ws, size_t ws_size,
                              hipStream_t stream) {
  const float* hs     = (const float*)d_in[0];
  /* d_in[1] = attention_mask: deterministic causal, recomputed in-kernel */
  const float* w_proj = (const float*)d_in[2];
  const float* w_o    = (const float*)d_in[3];
  float* out = (float*)d_out;

  float* red = (float*)d_ws;
  const size_t HEADSZ = (size_t)NH * SEQ * HD;  // 4,194,304 f16 elems
  _Float16* qg  = (_Float16*)((char*)d_ws + 256);
  _Float16* kg  = qg + HEADSZ;
  _Float16* vgT = kg + HEADSZ;
  _Float16* ctx = vgT + HEADSZ;

  init_red<<<1, 32, 0, stream>>>(red);
  absmean_reduce<<<512, 256, 0, stream>>>(w_proj, NOUT * HIDDEN, red + 0);
  absmean_reduce<<<512, 256, 0, stream>>>(w_o, HIDDEN * HIDDEN, red + 1);
  finalize_scales<<<1, 32, 0, stream>>>(red);

  dim3 g0(SEQ / 128, NOUT / 128);
  gemm_ws<0><<<g0, 256, 0, stream>>>(hs, nullptr, w_proj, red + 2, qg, kg, vgT, nullptr);

  rope_kernel<<<dim3(NH, SEQ), 64, 0, stream>>>(qg, kg);

  attn_kernel<<<dim3(NH, SEQ / 128), 256, 0, stream>>>(qg, kg, vgT, ctx);

  dim3 g1(SEQ / 128, HIDDEN / 128);
  gemm_ws<1><<<g1, 256, 0, stream>>>(nullptr, ctx, w_o, red + 4, nullptr, nullptr, nullptr, out);
}
